// SGC_Net_17617955848511
// MI455X (gfx1250) — compile-verified, matrix operand order checked by reading, not run on hardware
//
#include <hip/hip_runtime.h>

// ---------------------------------------------------------------------------
// SGC (Simple Graph Convolution), K=2 hops, on MI455X (gfx1250, wave32).
//
//   deg[i]  = 1 + #incoming edges          (self loops added)
//   dinv    = rsqrt(deg)
//   h1      = scatter_sum(norm * x[src])   (+ self loop dinv^2 * x)
//   h2      = scatter_sum(norm * h1[src])  (+ self loop dinv^2 * h1)
//   out     = log_softmax(h2 @ W^T + b)    -> WMMA bf16 path
//
// Propagation is memory/atomic bound and stays fp32 (atomics land in the
// 192MB L2 since both buffers are 51.2MB). The projection uses
// v_wmma_f32_16x16x32_bf16 with in-register fp32->bf16 conversion.
// ---------------------------------------------------------------------------

#define N_NODES 100000
#define N_EDGES 625000
#define F_IN    128
#define N_CLASS 32

typedef __attribute__((ext_vector_type(16))) __bf16 v16bf;
typedef __attribute__((ext_vector_type(8)))  float  v8f;

// deg[i] = 1.0f  (self loop contribution)
__global__ void init_deg_kernel(float* __restrict__ deg, int n) {
    int i = blockIdx.x * blockDim.x + threadIdx.x;
    if (i < n) deg[i] = 1.0f;
}

// deg[dst] += 1 per edge
__global__ void count_deg_kernel(const int* __restrict__ dst_idx,
                                 float* __restrict__ deg, int E) {
    int e = blockIdx.x * blockDim.x + threadIdx.x;
    if (e < E) atomicAdd(&deg[dst_idx[e]], 1.0f);
}

// deg -> rsqrt(deg) in place (deg >= 1 always due to self loops)
__global__ void rsqrt_kernel(float* __restrict__ deg, int n) {
    int i = blockIdx.x * blockDim.x + threadIdx.x;
    if (i < n) deg[i] = rsqrtf(deg[i]);
}

// yout[i][:] = dinv[i]^2 * xin[i][:]   (self-loop term, also initializes yout)
// one thread per float4: n * (F_IN/4) threads
__global__ void self_init_kernel(const float* __restrict__ xin,
                                 float* __restrict__ yout,
                                 const float* __restrict__ dinv, int n) {
    int idx = blockIdx.x * blockDim.x + threadIdx.x;   // float4 index
    if (idx >= n * (F_IN / 4)) return;
    int node = idx >> 5;                               // / (F_IN/4)
    float w = dinv[node];
    w *= w;
    float4 v = ((const float4*)xin)[idx];
    float4 r = {v.x * w, v.y * w, v.z * w, v.w * w};
    ((float4*)yout)[idx] = r;
}

// yout[dst][:] += dinv[src]*dinv[dst] * xin[src][:]
// one wave (32 lanes) per edge; each lane handles one float4 (4 features).
__global__ void prop_edges_kernel(const float* __restrict__ xin,
                                  float* __restrict__ yout,
                                  const float* __restrict__ dinv,
                                  const int* __restrict__ src_idx,
                                  const int* __restrict__ dst_idx, int E) {
    int tid  = blockIdx.x * blockDim.x + threadIdx.x;
    int e    = tid >> 5;            // wave-uniform (blockDim multiple of 32)
    if (e >= E) return;
    int lane = tid & 31;
    int src  = __builtin_amdgcn_readfirstlane(src_idx[e]);
    int dst  = __builtin_amdgcn_readfirstlane(dst_idx[e]);
    float w  = dinv[src] * dinv[dst];

    const float4 v = *(const float4*)(xin + (size_t)src * F_IN + lane * 4);
    float* o = yout + (size_t)dst * F_IN + lane * 4;
    atomicAdd(o + 0, v.x * w);
    atomicAdd(o + 1, v.y * w);
    atomicAdd(o + 2, v.z * w);
    atomicAdd(o + 3, v.w * w);
}

// One wave per 16-node tile: logits[16x32] = h[16x128] @ W[32x128]^T + b,
// then row-wise log_softmax. 8 x v_wmma_f32_16x16x32_bf16 per tile.
//
// 16-bit A-matrix 16x32 lane layout (ISA 7.12.2):
//   lane L<16 : row M=L,  halfs hold K = koff+{0..7} and koff+{16..23}
//   lane L>=16: row M=L-16, K = koff+{8..15} and koff+{24..31}
// B (32x16, K-major per lane) mirrors this with N = lane&15.
// C/D f32: VGPR r, lanes 0-15 -> (M=r, N=lane); lanes 16-31 -> (M=r+8).
__global__ void gemm_lsm_kernel(const float* __restrict__ h,
                                const float* __restrict__ W,
                                const float* __restrict__ bias,
                                float* __restrict__ out, int ntiles) {
    int wave = threadIdx.x >> 5;
    int lane = threadIdx.x & 31;
    int tile = blockIdx.x * (blockDim.x >> 5) + wave;
    if (tile >= ntiles) return;           // wave-uniform: EXEC all-ones inside

    int half  = lane >> 4;
    int idx16 = lane & 15;
    int node  = tile * 16 + idx16;        // A-matrix row for this lane
    int kbase = half * 8;

    const float* arow  = h + (size_t)node * F_IN;
    const float* b0row = W + (size_t)idx16 * F_IN;          // class idx16
    const float* b1row = W + (size_t)(idx16 + 16) * F_IN;   // class idx16+16

    v8f c0 = {};  // classes 0..15
    v8f c1 = {};  // classes 16..31

#pragma unroll
    for (int kk = 0; kk < 4; ++kk) {
        int koff = kk * 32 + kbase;
        v16bf a, b0, b1;
#pragma unroll
        for (int i = 0; i < 8; ++i) {
            a[i]      = (__bf16)arow[koff + i];
            a[i + 8]  = (__bf16)arow[koff + 16 + i];
            b0[i]     = (__bf16)b0row[koff + i];
            b0[i + 8] = (__bf16)b0row[koff + 16 + i];
            b1[i]     = (__bf16)b1row[koff + i];
            b1[i + 8] = (__bf16)b1row[koff + 16 + i];
        }
        c0 = __builtin_amdgcn_wmma_f32_16x16x32_bf16(false, a, false, b0,
                                                     (short)0, c0, false, false);
        c1 = __builtin_amdgcn_wmma_f32_16x16x32_bf16(false, a, false, b1,
                                                     (short)0, c1, false, false);
    }

    float bb0 = bias[idx16];
    float bb1 = bias[idx16 + 16];

#pragma unroll
    for (int r = 0; r < 8; ++r) {
        float l0 = c0[r] + bb0;           // (row r+8*half, class idx16)
        float l1 = c1[r] + bb1;           // (row r+8*half, class idx16+16)

        // reduce across the 16 lanes holding this row (masks stay in-half)
        float m = fmaxf(l0, l1);
#pragma unroll
        for (int msk = 1; msk < 16; msk <<= 1)
            m = fmaxf(m, __shfl_xor(m, msk, 32));
        float s = __expf(l0 - m) + __expf(l1 - m);
#pragma unroll
        for (int msk = 1; msk < 16; msk <<= 1)
            s += __shfl_xor(s, msk, 32);
        float lse = m + __logf(s);

        int mrow = r + half * 8;
        size_t o = (size_t)(tile * 16 + mrow) * N_CLASS;
        out[o + idx16]      = l0 - lse;
        out[o + idx16 + 16] = l1 - lse;
    }
}

extern "C" void kernel_launch(void* const* d_in, const int* in_sizes, int n_in,
                              void* d_out, int out_size, void* d_ws, size_t ws_size,
                              hipStream_t stream) {
    (void)in_sizes; (void)n_in; (void)out_size; (void)ws_size;

    const float* x    = (const float*)d_in[0];          // [N, 128]
    const int*   ei   = (const int*)d_in[1];            // [2, E] (harness: int)
    const float* W    = (const float*)d_in[2];          // [32, 128]
    const float* bias = (const float*)d_in[3];          // [32]
    float*       out  = (float*)d_out;                  // [N, 32]

    const int* src = ei;                // edge_index[0] = source j
    const int* dst = ei + N_EDGES;      // edge_index[1] = target i

    // Workspace: dinv | h1 | h2   (~103 MB total)
    char*  ws    = (char*)d_ws;
    float* dinv  = (float*)ws;
    size_t off   = ((size_t)N_NODES * sizeof(float) + 255) & ~(size_t)255;
    float* h1    = (float*)(ws + off);
    float* h2    = (float*)(ws + off + (size_t)N_NODES * F_IN * sizeof(float));

    const int T = 256;

    init_deg_kernel<<<(N_NODES + T - 1) / T, T, 0, stream>>>(dinv, N_NODES);
    count_deg_kernel<<<(N_EDGES + T - 1) / T, T, 0, stream>>>(dst, dinv, N_EDGES);
    rsqrt_kernel<<<(N_NODES + T - 1) / T, T, 0, stream>>>(dinv, N_NODES);

    // hop 1: x -> h1
    {
        int nvec = N_NODES * (F_IN / 4);
        self_init_kernel<<<(nvec + T - 1) / T, T, 0, stream>>>(x, h1, dinv, N_NODES);
        long long nth = (long long)N_EDGES * 32;
        prop_edges_kernel<<<(int)((nth + T - 1) / T), T, 0, stream>>>(
            x, h1, dinv, src, dst, N_EDGES);
    }
    // hop 2: h1 -> h2
    {
        int nvec = N_NODES * (F_IN / 4);
        self_init_kernel<<<(nvec + T - 1) / T, T, 0, stream>>>(h1, h2, dinv, N_NODES);
        long long nth = (long long)N_EDGES * 32;
        prop_edges_kernel<<<(int)((nth + T - 1) / T), T, 0, stream>>>(
            h1, h2, dinv, src, dst, N_EDGES);
    }

    // projection + log_softmax: 6250 tiles of 16 nodes, 8 waves per block
    int ntiles = N_NODES / 16;                     // 6250 exactly
    int blocks = (ntiles + 7) / 8;
    gemm_lsm_kernel<<<blocks, 256, 0, stream>>>(h2, W, bias, out, ntiles);
}
